// ContactGraspnet_61469571940627
// MI455X (gfx1250) — matrix operand order repaired and checked by reading, not compiled, not run on hardware
//
#include <hip/hip_runtime.h>
#include <hip/hip_bf16.h>
#include <stdint.h>

typedef _Float16 h16;
typedef _Float16 v8h  __attribute__((ext_vector_type(8)));
typedef _Float16 v16h __attribute__((ext_vector_type(16)));
typedef float    v8f  __attribute__((ext_vector_type(8)));
typedef unsigned int u32x4 __attribute__((ext_vector_type(4)));
typedef int      i32x8 __attribute__((ext_vector_type(8)));
typedef int      i32x4 __attribute__((ext_vector_type(4)));

#define BATCH 2
#define NPTS  20000

// ---------------------------------------------------------------- FPS
__global__ void fps_kernel(const float* __restrict__ xyz, int N, int npoint,
                           int* __restrict__ outIdx, float* __restrict__ distBuf) {
  const int b = blockIdx.x;
  const float* p = xyz + (size_t)b * N * 3;
  float* dist = distBuf + (size_t)b * N;
  int* oi = outIdx + (size_t)b * npoint;
  const int tid = threadIdx.x, T = blockDim.x;
  __shared__ float sv[256];
  __shared__ int   si[256];
  __shared__ float cx, cy, cz;
  __shared__ int   sfar;
  for (int i = tid; i < N; i += T) dist[i] = 1e10f;
  if (tid == 0) sfar = 0;
  __syncthreads();
  for (int it = 0; it < npoint; ++it) {
    if (tid == 0) {
      int far = sfar;
      oi[it] = far;
      cx = p[far*3+0]; cy = p[far*3+1]; cz = p[far*3+2];
    }
    __syncthreads();
    float bv = -1.f; int bi = 0;
    for (int i = tid; i < N; i += T) {
      float dx = p[i*3+0]-cx, dy = p[i*3+1]-cy, dz = p[i*3+2]-cz;
      float d = fminf(dist[i], dx*dx+dy*dy+dz*dz);
      dist[i] = d;
      if (d > bv) { bv = d; bi = i; }
    }
    sv[tid] = bv; si[tid] = bi;
    __syncthreads();
    for (int s = T/2; s > 0; s >>= 1) {
      if (tid < s) {
        if (sv[tid+s] > sv[tid] || (sv[tid+s] == sv[tid] && si[tid+s] < si[tid])) {
          sv[tid] = sv[tid+s]; si[tid] = si[tid+s];
        }
      }
      __syncthreads();
    }
    if (tid == 0) sfar = si[0];
    __syncthreads();
  }
}

// ---------------------------------------------------------------- gather xyz
__global__ void gather3_kernel(const float* __restrict__ xyz, const int* __restrict__ idx,
                               int N, int S, int total, float* __restrict__ out) {
  int t = blockIdx.x * blockDim.x + threadIdx.x;
  if (t >= total) return;
  int c = t % 3, bs = t / 3;
  int b = bs / S;
  out[t] = xyz[((size_t)b * N + idx[bs]) * 3 + c];
}

// ---------------------------------------------------------------- ball query (first-K ascending index)
__global__ void ball_query_kernel(const float* __restrict__ xyz_b, const float* __restrict__ nxyz_b,
                                  int N, int S, float r2, int K, int* __restrict__ gidx) {
  int s = blockIdx.x * blockDim.x + threadIdx.x;
  if (s >= S) return;
  float qx = nxyz_b[s*3+0], qy = nxyz_b[s*3+1], qz = nxyz_b[s*3+2];
  int base = s * K, cnt = 0;
  for (int i = 0; i < N && cnt < K; ++i) {
    float dx = xyz_b[i*3+0]-qx, dy = xyz_b[i*3+1]-qy, dz = xyz_b[i*3+2]-qz;
    if (dx*dx+dy*dy+dz*dz <= r2) gidx[base + cnt++] = i;
  }
  int f = (cnt > 0) ? gidx[base] : 0;
  for (; cnt < K; ++cnt) gidx[base + cnt] = f;
}

// ---------------------------------------------------------------- grouped A matrix: [feat | rel_xyz | pad]
__global__ void build_A_sa_kernel(const float* __restrict__ xyz_b, const float* __restrict__ feat_b,
                                  int featC, const float* __restrict__ nxyz_b,
                                  const int* __restrict__ gidx, int S, int K, int Kpad,
                                  h16* __restrict__ A, size_t total) {
  size_t t = (size_t)blockIdx.x * blockDim.x + threadIdx.x;
  if (t >= total) return;
  int c = (int)(t % Kpad);
  size_t row = t / Kpad;
  int s = (int)(row / K);
  int j = gidx[row];
  float v = 0.f;
  if (c < featC)            v = feat_b[(size_t)j * featC + c];
  else if (c < featC + 3)   v = xyz_b[j*3 + (c-featC)] - nxyz_b[s*3 + (c-featC)];
  A[t] = (h16)v;
}

// SA4 group-all: [xyz | feat | pad]
__global__ void build_A_sa4_kernel(const float* __restrict__ xyz_b, const float* __restrict__ pts_b,
                                   int S, int C, int Kpad, h16* __restrict__ A, size_t total) {
  size_t t = (size_t)blockIdx.x * blockDim.x + threadIdx.x;
  if (t >= total) return;
  int c = (int)(t % Kpad);
  size_t row = t / Kpad;
  float v = 0.f;
  if (c < 3)          v = xyz_b[row*3 + c];
  else if (c < 3 + C) v = pts_b[row*C + (c-3)];
  A[t] = (h16)v;
}

// FP3: [points1 | broadcast(l4)]
__global__ void build_A_fp3_kernel(const float* __restrict__ pts1, const float* __restrict__ l4,
                                   int S1, int C1, int C2, int Kpad, h16* __restrict__ A, size_t total) {
  size_t t = (size_t)blockIdx.x * blockDim.x + threadIdx.x;
  if (t >= total) return;
  int c = (int)(t % Kpad);
  size_t row = t / Kpad;
  int b = (int)(row / S1), s = (int)(row % S1);
  float v;
  if (c < C1) v = pts1[((size_t)b*S1 + s)*C1 + c];
  else        v = l4[(size_t)b*C2 + (c - C1)];
  A[t] = (h16)v;
}

// ---------------------------------------------------------------- 3-NN
__global__ void three_nn_kernel(const float* __restrict__ xyz1, const float* __restrict__ xyz2,
                                int S1, int S2, int total, int* __restrict__ nnidx, float* __restrict__ nnw) {
  int t = blockIdx.x * blockDim.x + threadIdx.x;
  if (t >= total) return;
  int b = t / S1, s = t % S1;
  const float* p1 = xyz1 + ((size_t)b*S1 + s)*3;
  const float* p2 = xyz2 + (size_t)b*S2*3;
  float x = p1[0], y = p1[1], z = p1[2];
  float d0 = 1e30f, d1 = 1e30f, d2 = 1e30f;
  int   i0 = 0, i1 = 0, i2 = 0;
  for (int i = 0; i < S2; ++i) {
    float dx = p2[i*3+0]-x, dy = p2[i*3+1]-y, dz = p2[i*3+2]-z;
    float d = dx*dx+dy*dy+dz*dz;
    if (d < d0)      { d2=d1;i2=i1; d1=d0;i1=i0; d0=d;i0=i; }
    else if (d < d1) { d2=d1;i2=i1; d1=d;i1=i; }
    else if (d < d2) { d2=d;i2=i; }
  }
  float w0 = 1.f/(d0+1e-8f), w1 = 1.f/(d1+1e-8f), w2 = 1.f/(d2+1e-8f);
  float ws = w0+w1+w2;
  size_t o = ((size_t)b*S1 + s)*3;
  nnidx[o+0]=i0; nnidx[o+1]=i1; nnidx[o+2]=i2;
  nnw[o+0]=w0/ws; nnw[o+1]=w1/ws; nnw[o+2]=w2/ws;
}

// FP: [points1 | interp(points2)]
__global__ void build_A_fp_kernel(const float* __restrict__ pts1, const h16* __restrict__ pts2,
                                  const int* __restrict__ nnidx, const float* __restrict__ nnw,
                                  int S1, int S2, int C1, int C2, int Kpad,
                                  h16* __restrict__ A, size_t total) {
  size_t t = (size_t)blockIdx.x * blockDim.x + threadIdx.x;
  if (t >= total) return;
  int c = (int)(t % Kpad);
  size_t row = t / Kpad;
  int b = (int)(row / S1), s = (int)(row % S1);
  float v;
  if (c < C1) v = pts1[((size_t)b*S1 + s)*C1 + c];
  else {
    int cc = c - C1;
    size_t o = ((size_t)b*S1 + s)*3;
    const h16* base = pts2 + (size_t)b*S2*C2;
    v = nnw[o+0]*(float)base[(size_t)nnidx[o+0]*C2+cc]
      + nnw[o+1]*(float)base[(size_t)nnidx[o+1]*C2+cc]
      + nnw[o+2]*(float)base[(size_t)nnidx[o+2]*C2+cc];
  }
  A[t] = (h16)v;
}

// ---------------------------------------------------------------- weight pack into per-lane B-fragment layout
// packed[((kt*Ntiles+u)*32+lane)*16+h] = W[kt*32 + 16*(lane>=16) + h][u*16 + (lane&15)]
__global__ void pack_weights_kernel(const float* __restrict__ W, int cin, int cout, int Kpad,
                                    h16* __restrict__ P, size_t total) {
  size_t t = (size_t)blockIdx.x * blockDim.x + threadIdx.x;
  if (t >= total) return;
  int Ntiles = cout >> 4;
  int h = (int)(t & 15);
  int lane = (int)((t >> 4) & 31);
  size_t tile = t >> 9;
  int u = (int)(tile % Ntiles);
  int tk = (int)(tile / Ntiles);
  int Ki = tk*32 + ((lane >= 16) ? 16 : 0) + h;
  int nc = u*16 + (lane & 15);
  float v = (Ki < cin) ? W[(size_t)Ki*cout + nc] : 0.f;
  P[t] = (h16)v;
}

// fold bias+BN into scale/shift
__global__ void prep_ss_kernel(const float* __restrict__ b, const float* __restrict__ beta,
                               const float* __restrict__ gamma, const float* __restrict__ mean,
                               const float* __restrict__ var, int cout,
                               float* __restrict__ scale, float* __restrict__ shift) {
  int c = blockIdx.x * blockDim.x + threadIdx.x;
  if (c >= cout) return;
  float s = gamma[c] * rsqrtf(var[c] + 1e-5f);
  scale[c] = s;
  shift[c] = (b[c] - mean[c]) * s + beta[c];
}

// ---------------------------------------------------------------- WMMA GEMM + BN + ReLU
// 256 threads = 8 waves: 4 M-tiles x 2 N-tiles (64x32 macro-tile).
// B panels double-buffered in LDS via TDM (2 descriptors in flight), 8 K-tiles/chunk.
// D = A(rows x Kpad) * W(Kpad x cout), fp32 accumulate, fused BN+ReLU, fp16 out.
__global__ void __launch_bounds__(256)
wmma_gemm_bn_relu(const h16* __restrict__ A, const h16* __restrict__ Bp,
                  const float* __restrict__ scale, const float* __restrict__ shift,
                  h16* __restrict__ out, int Kpad, int cout) {
  __shared__ __align__(16) h16 bsh[2][8 * 1024];   // 2 x 16KB double buffer
  const int lane  = threadIdx.x;                   // 0..31
  const int wid   = threadIdx.y;                   // 0..7
  const int mtile = wid & 3;
  const int nsub  = wid >> 2;                      // 0..1
  const int Ntiles = gridDim.y * 2;                // cout/16
  const int mbase = blockIdx.x * 64 + mtile * 16;
  const int u0    = blockIdx.y * 2;                // first N-tile of this block
  const int l15 = lane & 15;
  const int hi  = lane >> 4;
  const h16* arow = A + (size_t)(mbase + l15) * Kpad;
  v8f acc = {0.f,0.f,0.f,0.f,0.f,0.f,0.f,0.f};
  const int ktn = Kpad >> 5;
  const int nchunks = (ktn + 7) >> 3;
  const unsigned ldsbase = (unsigned)(uintptr_t)&bsh[0][0];

  // TDM issue: 2D descriptor, CH rows of 2048B (256 x 8B), row stride Ntiles*1024B
  auto issue = [&](int chunk) {
    const int c0 = chunk * 8;
    const int CH = (ktn - c0 < 8) ? (ktn - c0) : 8;
    unsigned long long ga = (unsigned long long)(uintptr_t)(Bp + ((size_t)c0 * Ntiles + u0) * 512);
    unsigned strideQW = (unsigned)Ntiles * 128u;   // in 8-byte units
    u32x4 g0;
    g0[0] = 1u;                                    // count=1, user mode
    g0[1] = ldsbase + (unsigned)(chunk & 1) * (8u * 1024u * 2u);  // lds_addr
    g0[2] = (unsigned)ga;                          // global_addr[31:0]
    g0[3] = (unsigned)((ga >> 32) & 0x1FFFFFFu) | (2u << 30);     // addr[56:32] | type=2
    i32x8 g1;
    g1[0] = (int)(3u << 16);                       // data_size = 8B
    g1[1] = (int)(256u << 16);                     // tensor_dim0[15:0] = 256 (bits 63:48)
    g1[2] = (int)((unsigned)CH << 16);             // tensor_dim1[15:0] = CH (bits 95:80)
    g1[3] = (int)(256u << 16);                     // tile_dim0 = 256 (bits 127:112)
    g1[4] = (int)CH;                               // tile_dim1 = CH (bits 143:128)
    g1[5] = (int)strideQW;                         // tensor_dim0_stride[31:0]
    g1[6] = 0;
    g1[7] = 0;
    i32x4 z4 = {0,0,0,0};
    i32x8 z8 = {0,0,0,0,0,0,0,0};
    __builtin_amdgcn_tensor_load_to_lds(g0, g1, z4, z4, z8, 0);
  };
  auto do_k = [&](int kt, const h16* bl) {
    const h16* ap = arow + kt*32 + hi*16;
    v8h a0 = *(const v8h*)ap;
    v8h a1 = *(const v8h*)(ap + 8);
    v8h b0 = *(const v8h*)bl;
    v8h b1 = *(const v8h*)(bl + 8);
    v16h af = __builtin_shufflevector(a0, a1, 0,1,2,3,4,5,6,7,8,9,10,11,12,13,14,15);
    v16h bf = __builtin_shufflevector(b0, b1, 0,1,2,3,4,5,6,7,8,9,10,11,12,13,14,15);
    acc = __builtin_amdgcn_wmma_f32_16x16x32_f16(false, af, false, bf, (short)0, acc, false, false);
  };

  if (wid == 0) issue(0);                          // prologue: chunk 0 in flight
  for (int ch = 0; ch < nchunks; ++ch) {
    if (wid == 0) {
      if (ch + 1 < nchunks) {                      // keep next chunk in flight
        issue(ch + 1);
        __builtin_amdgcn_s_wait_tensorcnt(1);      // oldest (chunk ch) landed
      } else {
        __builtin_amdgcn_s_wait_tensorcnt(0);
      }
    }
    __syncthreads();                               // panel ch visible to all waves
    const h16* bufp = &bsh[ch & 1][0];
    const int c0 = ch * 8;
    const int CH = (ktn - c0 < 8) ? (ktn - c0) : 8;
    if (c0 + 8 < ktn)                              // prefetch next A panel once per chunk
      __builtin_prefetch(arow + (c0 + 8) * 32, 0, 1);
    if (CH == 8) {
#pragma unroll
      for (int kk = 0; kk < 8; ++kk)
        do_k(c0 + kk, bufp + kk*1024 + nsub*512 + lane*16);
    } else {
      for (int kk = 0; kk < CH; ++kk)
        do_k(c0 + kk, bufp + kk*1024 + nsub*512 + lane*16);
    }
    __syncthreads();                               // buffer may be refilled next iter
  }
  const int ncol = (u0 + nsub)*16 + l15;
  const float sc = scale[ncol], sh = shift[ncol];
#pragma unroll
  for (int r = 0; r < 8; ++r) {
    int m = mbase + hi*8 + r;
    float v = acc[r]*sc + sh;
    v = fmaxf(v, 0.f);
    out[(size_t)m*cout + ncol] = (h16)v;
  }
}

// ---------------------------------------------------------------- maxpool over K samples
__global__ void maxpool_kernel(const h16* __restrict__ X, int S, int K, int C,
                               float* __restrict__ pooled, int Ctot, int coff, int total) {
  int t = blockIdx.x * blockDim.x + threadIdx.x;
  if (t >= total) return;
  int s = t / C, c = t % C;
  float m = -3.4e38f;
  for (int k = 0; k < K; ++k)
    m = fmaxf(m, (float)X[((size_t)s*K + k)*C + c]);
  pooled[(size_t)s*Ctot + coff + c] = m;
}

// ---------------------------------------------------------------- head final linear (small cout)
__global__ void head_out_kernel(const h16* __restrict__ feat, const float* __restrict__ W2,
                                const float* __restrict__ b2, int rows, int cout,
                                float* __restrict__ out) {
  int t = blockIdx.x * blockDim.x + threadIdx.x;
  if (t >= rows * cout) return;
  int r = t / cout, c = t % cout;
  float acc = b2[c];
  const h16* fr = feat + (size_t)r * 128;
  for (int k = 0; k < 128; ++k) acc += (float)fr[k] * W2[k*cout + c];
  out[t] = acc;
}

// ---------------------------------------------------------------- final grasp assembly
__global__ void assemble_kernel(const float* __restrict__ l1_xyz, const float* __restrict__ dirO,
                                const float* __restrict__ appO, const float* __restrict__ offO,
                                const float* __restrict__ segO, int S, int total,
                                float* __restrict__ outG, float* __restrict__ outS,
                                float* __restrict__ outP, float* __restrict__ outOff,
                                float* __restrict__ outLg) {
  int t = blockIdx.x * blockDim.x + threadIdx.x;
  if (t >= total) return;
  int b = t / S, s = t % S;
  const float binv[10] = {0.0039721766f,0.0119165277f,0.0198608765f,0.0278052253f,0.0357495740f,
                          0.0436939228f,0.0516382716f,0.0595826204f,0.0675269692f,0.0750000000f};
  float dx = dirO[t*3+0], dy = dirO[t*3+1], dz = dirO[t*3+2];
  float dn = fmaxf(sqrtf(dx*dx+dy*dy+dz*dz), 1e-12f);
  float nx = dx/dn, ny = dy/dn, nz = dz/dn;
  float ax = appO[t*3+0], ay = appO[t*3+1], az = appO[t*3+2];
  float dot = nx*ax + ny*ay + nz*az;
  float ox = ax - dot*nx, oy = ay - dot*ny, oz = az - dot*nz;
  float on = fmaxf(sqrtf(ox*ox+oy*oy+oz*oz), 1e-12f);
  ox /= on; oy /= on; oz /= on;
  const float* lg = offO + (size_t)t*10;
  int bi = 0; float bv = lg[0];
  for (int i = 1; i < 10; ++i) if (lg[i] > bv) { bv = lg[i]; bi = i; }
  float th = binv[bi];
  // y = cross(orth, raw_dir)
  float yx = oy*dz - oz*dy, yy = oz*dx - ox*dz, yz = ox*dy - oy*dx;
  float px = l1_xyz[t*3+0], py = l1_xyz[t*3+1], pz = l1_xyz[t*3+2];
  float tx = px + 0.5f*th*dx - 0.1034f*ox;
  float ty = py + 0.5f*th*dy - 0.1034f*oy;
  float tz = pz + 0.5f*th*dz - 0.1034f*oz;
  float* G = outG + (size_t)t*16;
  G[0]=dx;  G[1]=yx;  G[2]=ox;  G[3]=tx;
  G[4]=dy;  G[5]=yy;  G[6]=oy;  G[7]=ty;
  G[8]=dz;  G[9]=yz;  G[10]=oz; G[11]=tz;
  G[12]=0.f;G[13]=0.f;G[14]=0.f;G[15]=1.f;
  outS[t] = 1.f / (1.f + expf(-segO[t]));
  outP[t*3+0]=px; outP[t*3+1]=py; outP[t*3+2]=pz;
  outOff[t] = th;
  for (int c = 0; c < 10; ++c) outLg[((size_t)b*10 + c)*S + s] = lg[c];
}

// =================================================================
// host orchestration
// =================================================================
struct ConvBN { const float *W,*b,*beta,*gamma,*mean,*var; int cin,cout; };
struct HeadP  { const float *W2,*b2; ConvBN l1; int cout2; };
struct PL     { h16* packed; float* scale; float* shift; int Kpad, cout; };

extern "C" void kernel_launch(void* const* d_in, const int* in_sizes, int n_in,
                              void* d_out, int out_size, void* d_ws, size_t ws_size,
                              hipStream_t stream) {
  (void)in_sizes; (void)n_in; (void)out_size; (void)ws_size;
  const float* pc = (const float*)d_in[0];

  // ---- parse params (dict flattened in sorted-key order) ----
  int ii = 1;
  auto nextConv = [&](int cin, int cout) -> ConvBN {
    ConvBN c;
    c.W     = (const float*)d_in[ii++]; c.b    = (const float*)d_in[ii++];
    c.beta  = (const float*)d_in[ii++]; c.gamma= (const float*)d_in[ii++];
    c.mean  = (const float*)d_in[ii++]; c.var  = (const float*)d_in[ii++];
    c.cin = cin; c.cout = cout; return c;
  };
  auto nextHead = [&](int co) -> HeadP {
    HeadP h;
    h.W2 = (const float*)d_in[ii++]; h.b2 = (const float*)d_in[ii++];
    h.l1 = nextConv(128, 128); h.cout2 = co; return h;
  };
  HeadP appH = nextHead(3);
  HeadP dirH = nextHead(3);
  ConvBN fp1L[3] = { nextConv(448,128), nextConv(128,128), nextConv(128,128) };
  ConvBN fp2L[2] = { nextConv(896,256), nextConv(256,128) };
  ConvBN fp3L[2] = { nextConv(1664,256), nextConv(256,256) };
  HeadP offH = nextHead(10);
  const int sa1W[3][3] = {{32,32,64},{64,64,128},{64,96,128}};
  const int sa2W[3][3] = {{64,64,128},{128,128,256},{128,128,256}};
  ConvBN sa1L[3][3], sa2L[3][3], sa3L[3][3];
  for (int m = 0; m < 3; ++m) { int c = 6;   for (int l = 0; l < 3; ++l) { sa1L[m][l] = nextConv(c, sa1W[m][l]); c = sa1W[m][l]; } }
  for (int m = 0; m < 3; ++m) { int c = 323; for (int l = 0; l < 3; ++l) { sa2L[m][l] = nextConv(c, sa2W[m][l]); c = sa2W[m][l]; } }
  for (int m = 0; m < 3; ++m) { int c = 643; for (int l = 0; l < 3; ++l) { sa3L[m][l] = nextConv(c, sa2W[m][l]); c = sa2W[m][l]; } }
  ConvBN sa4L[3] = { nextConv(643,256), nextConv(256,512), nextConv(512,1024) };
  HeadP segH = nextHead(1);

  // ---- workspace carving ----
  size_t wsoff = 0;
  auto alloc = [&](size_t bytes) -> void* {
    wsoff = (wsoff + 255) & ~(size_t)255;
    void* p = (char*)d_ws + wsoff;
    wsoff += bytes;
    return p;
  };
  float* l1_xyz = (float*)alloc((size_t)BATCH*2048*3*4);
  float* l2_xyz = (float*)alloc((size_t)BATCH*512*3*4);
  float* l3_xyz = (float*)alloc((size_t)BATCH*128*3*4);
  float* l1_pts = (float*)alloc((size_t)BATCH*2048*320*4);
  float* l2_pts = (float*)alloc((size_t)BATCH*512*640*4);
  float* l3_pts = (float*)alloc((size_t)BATCH*128*640*4);
  float* l4_pts = (float*)alloc((size_t)BATCH*1024*4);
  h16* fp3_out  = (h16*)alloc((size_t)BATCH*128*256*2);
  h16* fp2_out  = (h16*)alloc((size_t)BATCH*512*128*2);
  h16* fp1_out  = (h16*)alloc((size_t)BATCH*2048*128*2);
  h16* headfeat = (h16*)alloc((size_t)BATCH*2048*128*2);
  float* dirO   = (float*)alloc((size_t)BATCH*2048*3*4);
  float* appO   = (float*)alloc((size_t)BATCH*2048*3*4);
  float* offO   = (float*)alloc((size_t)BATCH*2048*10*4);
  float* segO   = (float*)alloc((size_t)BATCH*2048*4);
  float* fpsDist= (float*)alloc((size_t)BATCH*NPTS*4);
  int* idx1 = (int*)alloc((size_t)BATCH*2048*4);
  int* idx2 = (int*)alloc((size_t)BATCH*512*4);
  int* idx3 = (int*)alloc((size_t)BATCH*128*4);
  int* gidx = (int*)alloc((size_t)2048*128*4);
  int* nnidx = (int*)alloc((size_t)BATCH*2048*3*4);
  float* nnw = (float*)alloc((size_t)BATCH*2048*3*4);
  h16* bufA = (h16*)alloc((size_t)34000000*2);
  h16* bufB = (h16*)alloc((size_t)34000000*2);

  // ---- prep (pack weights + fold BN) ----
  auto prep = [&](const ConvBN& L) -> PL {
    PL p; p.cout = L.cout; p.Kpad = ((L.cin + 31) / 32) * 32;
    size_t tot = (size_t)p.Kpad * L.cout;
    p.packed = (h16*)alloc(tot * 2);
    p.scale  = (float*)alloc(L.cout * 4);
    p.shift  = (float*)alloc(L.cout * 4);
    pack_weights_kernel<<<(unsigned)((tot+255)/256),256,0,stream>>>(L.W, L.cin, L.cout, p.Kpad, p.packed, tot);
    prep_ss_kernel<<<(unsigned)((L.cout+63)/64),64,0,stream>>>(L.b, L.beta, L.gamma, L.mean, L.var, L.cout, p.scale, p.shift);
    return p;
  };
  PL sa1P[3][3], sa2P[3][3], sa3P[3][3], sa4P[3], fp1P[3], fp2P[2], fp3P[2];
  PL appP, dirP, offP, segP;
  for (int m = 0; m < 3; ++m) for (int l = 0; l < 3; ++l) { sa1P[m][l]=prep(sa1L[m][l]); sa2P[m][l]=prep(sa2L[m][l]); sa3P[m][l]=prep(sa3L[m][l]); }
  for (int l = 0; l < 3; ++l) sa4P[l] = prep(sa4L[l]);
  for (int l = 0; l < 3; ++l) fp1P[l] = prep(fp1L[l]);
  for (int l = 0; l < 2; ++l) fp2P[l] = prep(fp2L[l]);
  for (int l = 0; l < 2; ++l) fp3P[l] = prep(fp3L[l]);
  appP = prep(appH.l1); dirP = prep(dirH.l1); offP = prep(offH.l1); segP = prep(segH.l1);

  // GEMM: grid = (rows/64, cout/32), block = (32,8) -> 8 waves per block
  auto gemm = [&](const PL& p, const h16* Ain, size_t rows, h16* outp) {
    dim3 g((unsigned)(rows / 64), (unsigned)(p.cout / 32));
    wmma_gemm_bn_relu<<<g, dim3(32,8), 0, stream>>>(Ain, p.packed, p.scale, p.shift, outp, p.Kpad, p.cout);
  };
  auto runMLP = [&](PL* pls, int nl, size_t rows, h16* a, h16* b, h16* finalOut) -> h16* {
    h16* in = a;
    for (int i = 0; i < nl; ++i) {
      h16* out = (i == nl-1 && finalOut) ? finalOut : ((in == a) ? b : a);
      gemm(pls[i], in, rows, out);
      in = out;
    }
    return in;
  };

  // ---- SA stages (MSG) ----
  auto runSA = [&](const float* xyzAll, const float* featAll, int featC, int N, int S,
                   const float* rads, const int* Ks, PL (*mlps)[3],
                   int* fpsIdx, float* newXyz, float* pooled, int Ctot) {
    fps_kernel<<<BATCH, 256, 0, stream>>>(xyzAll, N, S, fpsIdx, fpsDist);
    int tot3 = BATCH*S*3;
    gather3_kernel<<<(tot3+255)/256, 256, 0, stream>>>(xyzAll, fpsIdx, N, S, tot3, newXyz);
    for (int b = 0; b < BATCH; ++b) {
      const float* xb = xyzAll + (size_t)b*N*3;
      const float* fb = featAll + (size_t)b*N*featC;
      const float* nb = newXyz + (size_t)b*S*3;
      int coff = 0;
      for (int sc = 0; sc < 3; ++sc) {
        int K = Ks[sc];
        ball_query_kernel<<<(S+63)/64, 64, 0, stream>>>(xb, nb, N, S, rads[sc]*rads[sc], K, gidx);
        int Kpad = mlps[sc][0].Kpad;
        size_t rows = (size_t)S*K;
        size_t tot = rows * Kpad;
        build_A_sa_kernel<<<(unsigned)((tot+255)/256), 256, 0, stream>>>(xb, fb, featC, nb, gidx, S, K, Kpad, bufA, tot);
        h16* last = runMLP(mlps[sc], 3, rows, bufA, bufB, nullptr);
        int C = mlps[sc][2].cout;
        int pt = S*C;
        maxpool_kernel<<<(pt+255)/256, 256, 0, stream>>>(last, S, K, C, pooled + (size_t)b*S*Ctot, Ctot, coff, pt);
        coff += C;
      }
    }
  };
  const float r1[3] = {0.02f,0.04f,0.08f}; const int K1[3] = {32,64,128};
  const float r2[3] = {0.04f,0.08f,0.16f}; const int K2[3] = {64,64,128};
  const float r3[3] = {0.08f,0.16f,0.32f}; const int K3[3] = {64,64,128};
  runSA(pc,     pc,     3,   NPTS, 2048, r1, K1, sa1P, idx1, l1_xyz, l1_pts, 320);
  runSA(l1_xyz, l1_pts, 320, 2048, 512,  r2, K2, sa2P, idx2, l2_xyz, l2_pts, 640);
  runSA(l2_xyz, l2_pts, 640, 512,  128,  r3, K3, sa3P, idx3, l3_xyz, l3_pts, 640);

  // ---- SA4 (group all) ----
  for (int b = 0; b < BATCH; ++b) {
    size_t tot = (size_t)128*672;
    build_A_sa4_kernel<<<(unsigned)((tot+255)/256), 256, 0, stream>>>(
        l3_xyz + (size_t)b*128*3, l3_pts + (size_t)b*128*640, 128, 640, 672, bufA, tot);
    h16* last = runMLP(sa4P, 3, 128, bufA, bufB, nullptr);
    maxpool_kernel<<<(1024+255)/256, 256, 0, stream>>>(last, 1, 128, 1024, l4_pts + (size_t)b*1024, 1024, 0, 1024);
  }

  // ---- FP3 (broadcast interp) ----
  {
    size_t rows = (size_t)BATCH*128, tot = rows*1664;
    build_A_fp3_kernel<<<(unsigned)((tot+255)/256), 256, 0, stream>>>(l3_pts, l4_pts, 128, 640, 1024, 1664, bufA, tot);
    runMLP(fp3P, 2, rows, bufA, bufB, fp3_out);
  }
  // ---- FP2 ----
  {
    int tn = BATCH*512;
    three_nn_kernel<<<(tn+63)/64, 64, 0, stream>>>(l2_xyz, l3_xyz, 512, 128, tn, nnidx, nnw);
    size_t rows = (size_t)BATCH*512, tot = rows*896;
    build_A_fp_kernel<<<(unsigned)((tot+255)/256), 256, 0, stream>>>(l2_pts, fp3_out, nnidx, nnw, 512, 128, 640, 256, 896, bufA, tot);
    runMLP(fp2P, 2, rows, bufA, bufB, fp2_out);
  }
  // ---- FP1 ----
  {
    int tn = BATCH*2048;
    three_nn_kernel<<<(tn+63)/64, 64, 0, stream>>>(l1_xyz, l2_xyz, 2048, 512, tn, nnidx, nnw);
    size_t rows = (size_t)BATCH*2048, tot = rows*448;
    build_A_fp_kernel<<<(unsigned)((tot+255)/256), 256, 0, stream>>>(l1_pts, fp2_out, nnidx, nnw, 2048, 512, 320, 128, 448, bufA, tot);
    runMLP(fp1P, 3, rows, bufA, bufB, fp1_out);
  }

  // ---- heads ----
  size_t hrows = (size_t)BATCH*2048;
  auto runHead = [&](const HeadP& H, const PL& P, float* outb) {
    gemm(P, fp1_out, hrows, headfeat);
    int tot = (int)hrows * H.cout2;
    head_out_kernel<<<(tot+255)/256, 256, 0, stream>>>(headfeat, H.W2, H.b2, (int)hrows, H.cout2, outb);
  };
  runHead(dirH, dirP, dirO);
  runHead(appH, appP, appO);
  runHead(offH, offP, offO);
  runHead(segH, segP, segO);

  // ---- assemble outputs ----
  float* out = (float*)d_out;
  float* outG   = out;
  float* outS   = out + (size_t)BATCH*2048*16;
  float* outP   = outS + (size_t)BATCH*2048;
  float* outOff = outP + (size_t)BATCH*2048*3;
  float* outLg  = outOff + (size_t)BATCH*2048;
  int at = BATCH*2048;
  assemble_kernel<<<(at+63)/64, 64, 0, stream>>>(l1_xyz, dirO, appO, offO, segO, 2048, at,
                                                 outG, outS, outP, outOff, outLg);
}
